// CapsuleRewardHead_86320252715285
// MI455X (gfx1250) — matrix operand order repaired and checked by reading, not compiled, not run on hardware
//
#include <hip/hip_runtime.h>

typedef __bf16 bf16_t;
typedef __attribute__((ext_vector_type(16))) __bf16 v16bf;
typedef __attribute__((ext_vector_type(8)))  float   v8f;

typedef int v4i_t __attribute__((vector_size(16)));
typedef __attribute__((address_space(1))) v4i_t as1_v4i;   // global int4
typedef __attribute__((address_space(3))) v4i_t as3_v4i;   // LDS int4

#define HIDDEN 4096
#define NCOUT  128          // NUM_CAPS * CAP_DIM
#define ROWS_PER_BLOCK 64
#define PART_STRIDE 132     // padded row stride (floats) -> conflict-free LDS
#define PART_TILE   (16 * PART_STRIDE)          // 2112 floats per 16-row tile
#define PART_HALF   (4 * PART_TILE)             // 8448 floats per K-half
#define PART_BYTES  (2 * PART_HALF * 4)         // 67584 bytes
#define CAPS_STRIDE 2064                        // padded per-object stride (floats)
#define STAGE_BUF   32768                       // one W-stage buffer
#define STAGE_BYTES (2 * STAGE_BUF)             // double buffered (caps alias: 33024 fits)
#define SMEM_BYTES  (PART_BYTES + STAGE_BYTES)  // 133120

#if defined(__AMDGCN__) && __has_builtin(__builtin_amdgcn_global_load_async_to_lds_b128)
#define HAVE_ASYNC 1
#else
#define HAVE_ASYNC 0
#endif

#if HAVE_ASYNC
#if __has_builtin(__builtin_amdgcn_s_wait_asynccnt)
#define WAIT_ASYNC(n) __builtin_amdgcn_s_wait_asynccnt(n)
#else
#define WAIT_ASYNC(n) asm volatile("s_wait_asynccnt %0" ::"i"(n) : "memory")
#endif
#else
#define WAIT_ASYNC(n)
#endif

// ---------------------------------------------------------------------------
// Prep: pack W [4096,128] f32 into WMMA A-fragment order, bf16 hi + lo.
// Fragment (ks,ct): 32 lanes x 16 bf16. 16-bit A 16x32 layout (ISA 7.12.2):
//   lane l: M = l%16 (the c index), h = l>>4; elem j: v=j/2, p=j&1,
//   k_local = (v>=4 ? 16:0) + h*8 + (v&3)*2 + p
// ---------------------------------------------------------------------------
__global__ __launch_bounds__(256) void pack_w_kernel(
    const float* __restrict__ W, bf16_t* __restrict__ whi, bf16_t* __restrict__ wlo) {
  int t  = blockIdx.x * blockDim.x + threadIdx.x;   // 524288 threads
  int j  = t & 15;
  int l  = (t >> 4) & 31;
  int ct = (t >> 9) & 7;
  int ks = t >> 12;
  int m = l & 15;
  int h = l >> 4;
  int v = j >> 1, p = j & 1;
  int klocal = ((v >= 4) ? 16 : 0) + h * 8 + (v & 3) * 2 + p;
  float w = W[(ks * 32 + klocal) * NCOUT + (ct * 16 + m)];
  bf16_t hi = (bf16_t)w;
  bf16_t lo = (bf16_t)(w - (float)hi);
  whi[t] = hi;
  wlo[t] = lo;
}

// Stage one k-step's W fragments (both K-halves, hi+lo = 32KB) into an LDS
// buffer. Async global->LDS when available (no VGPR staging, ASYNCcnt), else
// synchronous load + ds_store fallback.
__device__ __forceinline__ void stage_batch(const bf16_t* __restrict__ whi,
                                            const bf16_t* __restrict__ wlo,
                                            char* buf, int t, int s) {
  int chunk = t >> 6;                       // = kw*2 + which
  int skw = chunk >> 1, which = chunk & 1;
  const char* src = (const char*)((which ? wlo : whi) +
                    (long)(skw * 64 + s) * 4096 + (t & 63) * 64);
  char* dst = buf + chunk * 8192 + (t & 63) * 128;
#if HAVE_ASYNC
#pragma unroll
  for (int q = 0; q < 8; ++q)
    __builtin_amdgcn_global_load_async_to_lds_b128(
        (as1_v4i*)(unsigned long long)(src + q * 16),
        (as3_v4i*)(dst + q * 16), 0, 0);
#else
  const uint4* s4 = (const uint4*)src;
  uint4*       d4 = (uint4*)dst;
#pragma unroll
  for (int q = 0; q < 8; ++q) d4[q] = s4[q];
#endif
}

// ---------------------------------------------------------------------------
// Main fused kernel: bf16x3 WMMA GEMM (split-K over wave pairs) + capsule
// routing, one block = 64 batch rows. Double-buffered async W staging.
// ---------------------------------------------------------------------------
__global__ __launch_bounds__(256) void caps_head_kernel(
    const float* __restrict__ x,
    const bf16_t* __restrict__ whi,
    const bf16_t* __restrict__ wlo,
    const float* __restrict__ b_lin,
    const float* __restrict__ caps_g,
    float* __restrict__ out) {
  extern __shared__ char smem[];
  float* partF = (float*)smem;                 // [2 khalf][4 tile][16 row][132]
  char*  stage = smem + PART_BYTES;            // 2 x 32KB W staging, later caps
  float* capsF = (float*)stage;                // [4 obj][8*256 (+16 pad)]

  const int t    = threadIdx.x;
  const int lane = t & 31;
  const int w    = t >> 5;          // wave 0..7
  const int kw   = w >> 2;          // K half
  const int ntl  = w & 3;           // batch tile within block
  const int h    = lane >> 4;
  const int n    = lane & 15;

  const int rowBase = blockIdx.x * ROWS_PER_BLOCK;
  const long xrow = (long)(rowBase + ntl * 16 + n) * HIDDEN;

  v8f acc[8] = {};

  stage_batch(whi, wlo, stage, t, 0);          // prologue: batch 0 -> buf 0

  for (int s = 0; s < 64; ++s) {
    if (s + 1 < 64)                            // overlap copy(s+1) with wmma(s)
      stage_batch(whi, wlo, stage + ((s + 1) & 1) * STAGE_BUF, t, s + 1);

    // x B-fragment: lane n = l%16 (batch col), K = (j>=8 ? 16 : 0) + h*8 + (j&7)
    const int kb = (kw * 64 + s) * 32 + h * 8;
    const float4* xp = (const float4*)(x + xrow + kb);
    float xv[16];
#pragma unroll
    for (int q = 0; q < 4; ++q) {
      float4 f = xp[(q & 1) + (q >> 1) * 4];
      xv[q * 4 + 0] = f.x; xv[q * 4 + 1] = f.y;
      xv[q * 4 + 2] = f.z; xv[q * 4 + 3] = f.w;
    }
    v16bf bhi, blo;
#pragma unroll
    for (int q = 0; q < 16; ++q) {
      bf16_t hi = (bf16_t)xv[q];
      bhi[q] = hi;
      blo[q] = (bf16_t)(xv[q] - (float)hi);
    }

    // batch s complete (in-order ASYNCcnt; allow batch s+1 in flight)
    if (s + 1 < 64) { WAIT_ASYNC(8); } else { WAIT_ASYNC(0); }
    __syncthreads();                           // publish batch s to all waves

    const char* whb = stage + (s & 1) * STAGE_BUF + (kw * 2 + 0) * 8192 + lane * 32;
    const char* wlb = stage + (s & 1) * STAGE_BUF + (kw * 2 + 1) * 8192 + lane * 32;
#pragma unroll
    for (int ct = 0; ct < 8; ++ct) {
      v16bf ah = *(const v16bf*)(whb + ct * 1024);
      v16bf al = *(const v16bf*)(wlb + ct * 1024);
      acc[ct] = __builtin_amdgcn_wmma_f32_16x16x32_bf16(false, ah, false, bhi,
                                                        (short)0, acc[ct], false, false);
      acc[ct] = __builtin_amdgcn_wmma_f32_16x16x32_bf16(false, ah, false, blo,
                                                        (short)0, acc[ct], false, false);
      acc[ct] = __builtin_amdgcn_wmma_f32_16x16x32_bf16(false, al, false, bhi,
                                                        (short)0, acc[ct], false, false);
    }
    __syncthreads();                           // readers done -> buffer reusable
  }

  // store partial primary: C/D layout lane l: col n=l%16, VGPR v -> row h*8+v
  {
    float* pb = partF + kw * PART_HALF + ntl * PART_TILE + n * PART_STRIDE;
#pragma unroll
    for (int ct = 0; ct < 8; ++ct)
#pragma unroll
      for (int vv = 0; vv < 8; ++vv)
        pb[ct * 16 + h * 8 + vv] = acc[ct][vv];
  }
  __syncthreads();

  // combine K-halves + bias into part0; copy out_caps into reused stage LDS
  {
    int row = t >> 2;                       // 0..63
    int cb  = (t & 3) * 32;
    float* p0 = partF + (row >> 4) * PART_TILE + (row & 15) * PART_STRIDE + cb;
    const float* p1 = p0 + PART_HALF;
#pragma unroll
    for (int q = 0; q < 8; ++q) {
      float4 a  = *(float4*)(p0 + q * 4);
      float4 b4 = *(const float4*)(p1 + q * 4);
      float4 bl = *(const float4*)(b_lin + cb + q * 4);
      a.x += b4.x + bl.x; a.y += b4.y + bl.y;
      a.z += b4.z + bl.z; a.w += b4.w + bl.w;
      *(float4*)(p0 + q * 4) = a;
    }
#pragma unroll
    for (int o = 0; o < 4; ++o) {
      float4*       cd = (float4*)(capsF + o * CAPS_STRIDE) + t * 2;
      const float4* cs = (const float4*)(caps_g + o * 2048) + t * 2;
      cd[0] = cs[0]; cd[1] = cs[1];
    }
  }
  __syncthreads();

  // dynamic routing: 4 lanes per row; lane q = l>>3 owns object o=q (8 caps)
  {
    int q  = lane >> 3;
    int rl = (w << 3) | (lane & 7);         // row_local 0..63
    const float* prim = partF + (rl >> 4) * PART_TILE + (rl & 15) * PART_STRIDE;
    const float* cw0  = capsF + q * CAPS_STRIDE;
    float bb[8];
#pragma unroll
    for (int i = 0; i < 8; ++i) bb[i] = 0.f;
    float v[16];
#pragma unroll
    for (int j = 0; j < 16; ++j) v[j] = 0.f;

    for (int r = 0; r < 3; ++r) {
      float sa[16];
      float Z = 0.f;
#pragma unroll
      for (int j = 0; j < 16; ++j) sa[j] = 0.f;
      for (int i = 0; i < 8; ++i) {
        const float* cw = cw0 + i * 256;
        float u[16];
#pragma unroll
        for (int j = 0; j < 16; ++j) u[j] = 0.f;
#pragma unroll
        for (int c = 0; c < 16; ++c) {
          float f = prim[i * 16 + c];
          const float* cwc = cw + c * 16;
#pragma unroll
          for (int j = 0; j < 16; ++j) u[j] += f * cwc[j];
        }
        float a = 0.f;
#pragma unroll
        for (int j = 0; j < 16; ++j) a += u[j] * v[j];
        bb[i] += a;                          // b += agreement (0 on pass 0)
        float e = __expf(bb[i]);             // unnormalized softmax weight
        Z += e;
#pragma unroll
        for (int j = 0; j < 16; ++j) sa[j] += e * u[j];
      }
#pragma unroll
      for (int m = 8; m <= 16; m <<= 1) {
        Z += __shfl_xor(Z, m, 32);
#pragma unroll
        for (int j = 0; j < 16; ++j) sa[j] += __shfl_xor(sa[j], m, 32);
      }
      float rz = 1.f / Z;
      float n2 = 0.f;
      float sv[16];
#pragma unroll
      for (int j = 0; j < 16; ++j) { sv[j] = sa[j] * rz; n2 += sv[j] * sv[j]; }
      float nr = sqrtf(n2);
      float scale = (n2 / (1.f + n2)) / (nr + 1e-8f);
#pragma unroll
      for (int j = 0; j < 16; ++j) v[j] = scale * sv[j];
    }
    if (q == 0) {
      float n2 = 0.f;
#pragma unroll
      for (int j = 0; j < 16; ++j) n2 += v[j] * v[j];
      out[rowBase + rl] = sqrtf(n2);
    }
  }
}

extern "C" void kernel_launch(void* const* d_in, const int* in_sizes, int n_in,
                              void* d_out, int out_size, void* d_ws, size_t ws_size,
                              hipStream_t stream) {
  const float* x     = (const float*)d_in[0];
  const float* W     = (const float*)d_in[1];
  const float* b_lin = (const float*)d_in[2];
  const float* caps  = (const float*)d_in[3];
  float* out = (float*)d_out;

  bf16_t* whi = (bf16_t*)d_ws;          // 1 MB
  bf16_t* wlo = whi + 524288;           // 1 MB

  pack_w_kernel<<<2048, 256, 0, stream>>>(W, whi, wlo);

  (void)hipFuncSetAttribute((const void*)caps_head_kernel,
                            hipFuncAttributeMaxDynamicSharedMemorySize,
                            (int)SMEM_BYTES);
  caps_head_kernel<<<256, 256, SMEM_BYTES, stream>>>(x, whi, wlo, b_lin, caps, out);
}
